// ImprovedSuperPointContrastiveLoss3_7146825581132
// MI455X (gfx1250) — matrix operand order repaired and checked by reading, not compiled, not run on hardware
//
#include <hip/hip_runtime.h>
#include <hip/hip_bf16.h>
#include <hip/hip_fp16.h>

typedef _Float16 v16h __attribute__((ext_vector_type(16)));
typedef _Float16 v8h  __attribute__((ext_vector_type(8)));
typedef float    v8f  __attribute__((ext_vector_type(8)));

#define DIM 256
#define NCLS 13
#define TEMP_INV (1.0f / 0.07f)
#define ROWS_PER_BLOCK 64          // 4 waves x 16 rows
#define BLOCK_THREADS 128
#define SP_CHUNK 128               // superpoints staged in LDS per iteration
#define LDS_STRIDE 264             // 256 halfs + 8 pad (16B) against bank conflicts

// ---------------- init: zero histogram + output scalar ----------------
__global__ void init_kernel(int* counts, float* out, int m) {
    int gid = blockIdx.x * blockDim.x + threadIdx.x;
    if (gid < m * NCLS) counts[gid] = 0;
    if (gid == 0) *out = 0.0f;
}

// ---------------- histogram: counts[idx[i], label[i]] += 1 ----------------
__global__ void hist_kernel(const int* __restrict__ r2s, const int* __restrict__ lbl,
                            int* counts, int n) {
    int i = blockIdx.x * blockDim.x + threadIdx.x;
    if (i < n) atomicAdd(&counts[r2s[i] * NCLS + lbl[i]], 1);
}

// ---------------- superpoint prep: argmax label + L2-normalize -> f16 ----------------
// grid = M blocks of 32 threads (one wave per superpoint row)
__global__ void sp_prep_kernel(const float* __restrict__ sp, const int* __restrict__ counts,
                               _Float16* sp_h, int* sp_lbl) {
    const int row  = blockIdx.x;
    const int lane = threadIdx.x;

    if (lane == 0) {
        int best = 0, bc = counts[row * NCLS];
        for (int c = 1; c < NCLS; ++c) {
            int v = counts[row * NCLS + c];
            if (v > bc) { bc = v; best = c; }   // '>' keeps first max, matching argmax
        }
        sp_lbl[row] = best;
    }

    const float4* src = (const float4*)(sp + (size_t)row * DIM);
    float4 a = src[lane * 2];
    float4 b = src[lane * 2 + 1];
    float ss = a.x * a.x + a.y * a.y + a.z * a.z + a.w * a.w
             + b.x * b.x + b.y * b.y + b.z * b.z + b.w * b.w;
#pragma unroll
    for (int msk = 16; msk >= 1; msk >>= 1) ss += __shfl_xor(ss, msk, 32);
    float inv = rsqrtf(ss + 1e-12f);

    v8h o;
    o[0] = (_Float16)(a.x * inv); o[1] = (_Float16)(a.y * inv);
    o[2] = (_Float16)(a.z * inv); o[3] = (_Float16)(a.w * inv);
    o[4] = (_Float16)(b.x * inv); o[5] = (_Float16)(b.y * inv);
    o[6] = (_Float16)(b.z * inv); o[7] = (_Float16)(b.w * inv);
    *(v8h*)(sp_h + (size_t)row * DIM + lane * 8) = o;
}

// ---------------- fused main: normalize A rows, WMMA sweep, masked exp-sum, loss ----------------
__global__ __launch_bounds__(BLOCK_THREADS)
void ssl_main_kernel(const float* __restrict__ rp, const int* __restrict__ r2s,
                     const _Float16* __restrict__ sp_h, const int* __restrict__ sp_lbl,
                     float* out, int n, int m) {
    extern __shared__ char smem[];
    _Float16* lda  = (_Float16*)smem;                                              // 64 x 264 halfs
    _Float16* ldb  = (_Float16*)(smem + ROWS_PER_BLOCK * LDS_STRIDE * 2);          // 128 x 264 halfs
    int*      llbl = (int*)(smem + (ROWS_PER_BLOCK + SP_CHUNK) * LDS_STRIDE * 2);  // 128 ints
    float*    bsum = (float*)(llbl + SP_CHUNK);

    const int tid  = threadIdx.x;
    const int wave = tid >> 5;
    const int lane = tid & 31;
    const int r16  = lane & 15;       // col within tile (B/C/D lane striping)
    const int h    = lane >> 4;       // half of the wave
    const int row0 = blockIdx.x * ROWS_PER_BLOCK;

    // ---- Stage A: load raw rows, L2-normalize, store f16 into LDS ----
    for (int rr = 0; rr < 16; ++rr) {
        const int r = wave * 16 + rr;
        const float4* src = (const float4*)(rp + (size_t)(row0 + r) * DIM);
        float4 a = src[lane * 2];
        float4 b = src[lane * 2 + 1];
        float ss = a.x * a.x + a.y * a.y + a.z * a.z + a.w * a.w
                 + b.x * b.x + b.y * b.y + b.z * b.z + b.w * b.w;
#pragma unroll
        for (int msk = 16; msk >= 1; msk >>= 1) ss += __shfl_xor(ss, msk, 32);
        float inv = rsqrtf(ss + 1e-12f);
        v8h o;
        o[0] = (_Float16)(a.x * inv); o[1] = (_Float16)(a.y * inv);
        o[2] = (_Float16)(a.z * inv); o[3] = (_Float16)(a.w * inv);
        o[4] = (_Float16)(b.x * inv); o[5] = (_Float16)(b.y * inv);
        o[6] = (_Float16)(b.z * inv); o[7] = (_Float16)(b.w * inv);
        *(v8h*)(lda + r * LDS_STRIDE + lane * 8) = o;
    }
    __syncthreads();

    // ---- A fragments for all of K=256, kept in registers, reused for every col tile ----
    // 16-bit A 16x32 layout (ISA 7.12.2): lane (r16,h); e<8 -> K = k0+8h+e ; e>=8 -> K = k0+16+8h+(e-8)
    v16h afrag[8];
    {
        const _Float16* arow = lda + (wave * 16 + r16) * LDS_STRIDE;
#pragma unroll
        for (int kc = 0; kc < 8; ++kc) {
            const int k0 = kc * 32;
            v8h lo = *(const v8h*)(arow + k0 + 8 * h);
            v8h hi = *(const v8h*)(arow + k0 + 16 + 8 * h);
            afrag[kc] = __builtin_shufflevector(lo, hi, 0, 1, 2, 3, 4, 5, 6, 7,
                                                8, 9, 10, 11, 12, 13, 14, 15);
        }
    }

    // ---- per-lane row metadata: this lane's 8 C rows are m = v + 8*h ----
    int ownidx[8], plabel[8];
#pragma unroll
    for (int v = 0; v < 8; ++v) {
        const int gi = row0 + wave * 16 + v + 8 * h;
        const int ix = r2s[gi];
        ownidx[v] = ix;
        plabel[v] = sp_lbl[ix];
    }
    float pacc[8], nacc[8];
#pragma unroll
    for (int v = 0; v < 8; ++v) { pacc[v] = 0.0f; nacc[v] = 0.0f; }

    const int nchunk = m / SP_CHUNK;
    for (int ch = 0; ch < nchunk; ++ch) {
        __syncthreads();  // previous chunk's ldb reads done
        {
            const uint4* src = (const uint4*)(sp_h + (size_t)ch * SP_CHUNK * DIM);
            for (int u = tid; u < SP_CHUNK * DIM / 8; u += BLOCK_THREADS) {
                const int srow = u >> 5;   // DIM/8 = 32 uint4 per row
                const int sc   = u & 31;
                *(uint4*)(ldb + srow * LDS_STRIDE + sc * 8) = src[u];
            }
            if (tid < SP_CHUNK) llbl[tid] = sp_lbl[ch * SP_CHUNK + tid];
        }
        __syncthreads();

#pragma unroll 1
        for (int t = 0; t < SP_CHUNK / 16; ++t) {
            // hoist scalar/column metadata so the ds_load_b32 latency hides under WMMAs
            const int j    = ch * SP_CHUNK + t * 16 + r16;   // global superpoint col
            const int slbl = llbl[t * 16 + r16];

            v8f c = {};
            // 16-bit B 32x16 layout: lane col = r16; e<8 -> K = k0+8h+e ; e>=8 -> K = k0+16+8h+(e-8)
            const _Float16* brow = ldb + (t * 16 + r16) * LDS_STRIDE;
#pragma unroll
            for (int kc = 0; kc < 8; ++kc) {
                v8h lo = *(const v8h*)(brow + kc * 32 + 8 * h);
                v8h hi = *(const v8h*)(brow + kc * 32 + 16 + 8 * h);
                v16h bfrag = __builtin_shufflevector(lo, hi, 0, 1, 2, 3, 4, 5, 6, 7,
                                                     8, 9, 10, 11, 12, 13, 14, 15);
                c = __builtin_amdgcn_wmma_f32_16x16x32_f16(
                        false, afrag[kc], false, bfrag, (short)0, c, false, false);
            }

            // Branchless epilogue: unconditional v_exp_f32 (TRANS, co-executes with XDL),
            // masks lowered to v_cndmask instead of exec-divergent blocks.
#pragma unroll
            for (int v = 0; v < 8; ++v) {
                const float logit  = c[v] * TEMP_INV;
                const float e      = __expf(logit);
                const bool  own    = (ownidx[v] == j);
                const bool  masked = own | (plabel[v] == slbl);
                pacc[v] += own    ? logit : 0.0f;   // positive logit captured exactly once
                nacc[v] += masked ? 0.0f  : e;
            }
        }
    }

    // ---- reduce across the 16 lanes of each half (h bit untouched by masks < 16) ----
#pragma unroll
    for (int msk = 1; msk < 16; msk <<= 1) {
#pragma unroll
        for (int v = 0; v < 8; ++v) {
            nacc[v] += __shfl_xor(nacc[v], msk, 32);
            pacc[v] += __shfl_xor(pacc[v], msk, 32);
        }
    }

    float lsum = 0.0f;
    if (r16 == 0) {   // lane 0 (rows 0..7) and lane 16 (rows 8..15) of each wave
#pragma unroll
        for (int v = 0; v < 8; ++v) {
            const float ep = __expf(pacc[v]);
            lsum += -__logf(ep / (ep + nacc[v] + 1e-8f));
        }
    }
    __syncthreads();
    if (tid == 0) *bsum = 0.0f;
    __syncthreads();
    if (r16 == 0) atomicAdd(bsum, lsum);
    __syncthreads();
    if (tid == 0) atomicAdd(out, (*bsum) * (0.7f * 0.1f / (float)n));
}

extern "C" void kernel_launch(void* const* d_in, const int* in_sizes, int n_in,
                              void* d_out, int out_size, void* d_ws, size_t ws_size,
                              hipStream_t stream) {
    const float* sp_f = (const float*)d_in[0];   // superPoint_feat [M, 256]
    const float* rp_f = (const float*)d_in[1];   // rawPoint_feat   [N, 256]
    const int*   r2s  = (const int*)d_in[2];     // raw_to_super_index [N]
    const int*   lbl  = (const int*)d_in[3];     // label_inds [N]
    float*       out  = (float*)d_out;

    const int m = in_sizes[0] / DIM;             // 1024
    const int n = in_sizes[2];                   // 131072

    // workspace layout
    char* ws = (char*)d_ws;
    _Float16* sp_h   = (_Float16*)ws;                              // M*256 f16
    int*      sp_lab = (int*)(ws + (size_t)m * DIM * 2);           // M ints
    int*      counts = (int*)(ws + (size_t)m * DIM * 2 + m * 4);   // M*13 ints

    init_kernel<<<(m * NCLS + 255) / 256, 256, 0, stream>>>(counts, out, m);
    hist_kernel<<<(n + 255) / 256, 256, 0, stream>>>(r2s, lbl, counts, n);
    sp_prep_kernel<<<m, 32, 0, stream>>>(sp_f, counts, sp_h, sp_lab);

    const int smem_bytes = (ROWS_PER_BLOCK + SP_CHUNK) * LDS_STRIDE * 2
                         + SP_CHUNK * 4 + 16;
    hipFuncSetAttribute((const void*)ssl_main_kernel,
                        hipFuncAttributeMaxDynamicSharedMemorySize, smem_bytes);
    ssl_main_kernel<<<n / ROWS_PER_BLOCK, BLOCK_THREADS, smem_bytes, stream>>>(
        rp_f, r2s, sp_h, sp_lab, out, n, m);
}
